// SelfAttention_75170517615067
// MI455X (gfx1250) — matrix operand order repaired and checked by reading, not compiled, not run on hardware
//
#include <hip/hip_runtime.h>
#include <hip/hip_bf16.h>

// ---------------------------------------------------------------------------
// Causal self-attention, MI455X (gfx1250, wave32, WMMA, async-to-LDS).
//  pack_w   : swizzle Wq/Wk/Wv (f32) into WMMA B-fragment layout (f16 pairs).
//  qkv_proj : fused Q/K/V projection. X read ONCE (33.5 MB = HBM floor).
//             X tiles AND packed-W slices staged via
//             global_load_async_to_lds_b128, double-buffered, one barrier per
//             k-step; W slice shared by all 4 waves (4x L2-traffic cut, and
//             B-fragment reads become short-latency ds_load_b128).
//             V written pre-swizzled into B-fragment order for phase 2.
//  flash    : 16-query tile / wave, 32-key blocks, online softmax via
//             ds_swizzle row reductions, P relayout via LDS (ds_load_b128).
// ---------------------------------------------------------------------------

#define BB     2
#define SS     4096
#define DMODEL 1024
#define DK     64

typedef __attribute__((ext_vector_type(16))) _Float16     v16h;
typedef __attribute__((ext_vector_type(8)))  float        v8f;
typedef __attribute__((ext_vector_type(8)))  unsigned int v8u;

#define WMMA_F16(A, B, C) \
  __builtin_amdgcn_wmma_f32_16x16x32_f16(false, (A), false, (B), (short)0, (C), false, false)

// ---- CDNA5 async DMA: global -> LDS, 16 bytes per lane, tracked by ASYNCcnt
__device__ __forceinline__ void async_b128(unsigned ldsoff, const void* gaddr) {
  asm volatile("global_load_async_to_lds_b128 %0, %1, off"
               :: "v"(ldsoff), "v"(gaddr) : "memory");
}

// ---- ds_swizzle group-of-32 XOR reductions (rows striped across 16 lanes)
template <int IMM>
__device__ __forceinline__ float swz(float x) {
  return __builtin_bit_cast(
      float, __builtin_amdgcn_ds_swizzle(__builtin_bit_cast(int, x), IMM));
}
__device__ __forceinline__ float rowmax16(float v) {
  v = fmaxf(v, swz<0x041F>(v)); v = fmaxf(v, swz<0x081F>(v));
  v = fmaxf(v, swz<0x101F>(v)); v = fmaxf(v, swz<0x201F>(v));
  return v;
}
__device__ __forceinline__ float rowsum16(float v) {
  v += swz<0x041F>(v); v += swz<0x081F>(v);
  v += swz<0x101F>(v); v += swz<0x201F>(v);
  return v;
}

__device__ __forceinline__ v16h join16(uint4 lo, uint4 hi) {
  v8u t;
  t[0] = lo.x; t[1] = lo.y; t[2] = lo.z; t[3] = lo.w;
  t[4] = hi.x; t[5] = hi.y; t[6] = hi.z; t[7] = hi.w;
  return __builtin_bit_cast(v16h, t);
}

// ---------------------------------------------------------------------------
// Prologue: pack W into B-fragment layout.
// Wp[mat][kstep][nt][lane][v] (u32 = f16 pair {K, K+1}).  3*32768 u32.
// ---------------------------------------------------------------------------
__global__ void __launch_bounds__(256) pack_w_kernel(
    const float* __restrict__ Wq, const float* __restrict__ Wk,
    const float* __restrict__ Wv, unsigned* __restrict__ Wp) {
  const int e = blockIdx.x * 256 + threadIdx.x;      // 0 .. 3*32768-1
  const int mat = e >> 15;
  const int r   = e & 32767;                          // [ks:5][nt:2][lane:5][v:3]
  const float* W = (mat == 0) ? Wq : (mat == 1) ? Wk : Wv;
  const int v    = r & 7;
  const int lane = (r >> 3) & 31;
  const int nt   = (r >> 8) & 3;
  const int ks   = r >> 10;
  const int half = lane >> 4;
  const int n    = nt * 16 + (lane & 15);
  const int k    = ks * 32 + (v >> 2) * 16 + half * 8 + (v & 3) * 2;
  const unsigned pa = __builtin_bit_cast(unsigned short, (_Float16)W[(size_t)k * DK + n]);
  const unsigned pb = __builtin_bit_cast(unsigned short, (_Float16)W[(size_t)(k + 1) * DK + n]);
  Wp[e] = pa | (pb << 16);
}

// ---------------------------------------------------------------------------
// Phase 1: fused Q/K/V projection (X read once).
// Workgroup = 4 waves x 16 rows.  X tiles per-wave, W slice block-shared;
// both double-buffered via async DMA, one barrier per k-step.
// ---------------------------------------------------------------------------
__global__ void __launch_bounds__(128) qkv_proj_kernel(
    const float* __restrict__ X, const unsigned* __restrict__ Wp,
    _Float16* __restrict__ Qh, _Float16* __restrict__ Kh,
    _Float16* __restrict__ VpH /* u16 view of packed V */) {
  __shared__ float    xtile[4][2][16][32];     // 16 KB (per-wave A tiles)
  __shared__ unsigned wtile[2][12 * 32 * 8];   // 24 KB (block-shared B slices)

  const int tid  = threadIdx.x;
  const int wave = tid >> 5;
  const int lane = tid & 31;
  const int m0   = (blockIdx.x * 4 + wave) * 16;   // row tile over B*S
  const int M    = lane & 15;
  const int half = lane >> 4;

  v8f acc[12] = {};

  // async-copy one 16x32 f32 X tile (2 KB): 4 x 16B chunks per lane, coalesced
  auto issue_x = [&](int k0, int buf) {
    const unsigned xb = (unsigned)(uintptr_t)&xtile[wave][buf][0][0];
#pragma unroll
    for (int i = 0; i < 4; ++i) {
      const int ch  = i * 32 + lane;                 // 0..127 chunk of 16B
      const int row = ch >> 3;
      const int col = (ch & 7) * 4;                  // floats
      async_b128(xb + ch * 16,
                 (const void*)(X + (size_t)(m0 + row) * DMODEL + k0 + col));
    }
  };
  // async-copy one packed-W k-slice (12 KB, all 3 matrices): 6 chunks/thread
  auto issue_w = [&](int ks, int buf) {
    const unsigned wb = (unsigned)(uintptr_t)&wtile[buf][0];
    const unsigned* src = Wp + ks * 1024;
#pragma unroll
    for (int i = 0; i < 6; ++i) {
      const int ch  = i * 128 + tid;                 // 0..767 chunk of 16B
      const int mat = ch >> 8;                       // 256 chunks (4KB) per mat
      const int rem = ch & 255;
      async_b128(wb + ch * 16, (const void*)(src + mat * 32768 + rem * 4));
    }
  };

  issue_x(0, 0);
  issue_w(0, 0);
  asm volatile("s_wait_asynccnt 0x0" ::: "memory");
  __syncthreads();

  for (int k0 = 0; k0 < DMODEL; k0 += 32) {
    const int  cur  = (k0 >> 5) & 1;
    const bool more = (k0 + 32 < DMODEL);
    if (more) {                      // overlap next-step DMA with this compute
      issue_x(k0 + 32, cur ^ 1);
      issue_w((k0 >> 5) + 1, cur ^ 1);
    }

    // A fragment (16x32 f16) from LDS: VGPR v holds K = g*16 + half*8 + w*2 (+1)
    const float* xt = &xtile[wave][cur][0][0];
    v16h a;
#pragma unroll
    for (int v = 0; v < 8; ++v) {
      const int kp = ((v >> 2) << 4) + (half << 3) + ((v & 3) << 1);
      const float2 p = *(const float2*)(const void*)(xt + M * 32 + kp);
      a[2 * v]     = (_Float16)p.x;
      a[2 * v + 1] = (_Float16)p.y;
    }

    // 12 B fragments from block-shared LDS (short-latency ds_load_b128)
    const unsigned* wt = &wtile[cur][0] + lane * 8;
#pragma unroll
    for (int i = 0; i < 12; ++i) {
      const v16h bf = *(const v16h*)(const void*)(wt + i * 256);
      acc[i] = WMMA_F16(a, bf, acc[i]);
    }

    if (more) {
      asm volatile("s_wait_asynccnt 0x0" ::: "memory");  // next buffers landed
      __syncthreads();  // all waves' reads done + all waves' DMA visible
    }
  }

  // C layout: VGPR r -> row m0 + r + 8*half, col nt*16 + M
#pragma unroll
  for (int nt = 0; nt < 4; ++nt)
#pragma unroll
    for (int r = 0; r < 8; ++r) {
      const int row = m0 + r + (half << 3);
      const int col = (nt << 4) + M;
      Qh[(size_t)row * DK + col] = (_Float16)(acc[nt][r] * 0.125f); // 1/sqrt(64)
      Kh[(size_t)row * DK + col] = (_Float16)(acc[4 + nt][r]);
      // V scatter-stored directly in B-fragment order for phase 2:
      const int kb = row & 31, blk = row >> 5;
      const size_t u16i =
          (((((size_t)blk * 4 + nt) * 32) + ((kb >> 4) << 4) + M) * 8 +
           ((kb & 15) >> 1)) * 2 + (kb & 1);
      VpH[u16i] = (_Float16)(acc[8 + nt][r]);
    }
}

// ---------------------------------------------------------------------------
// Phase 2: flash attention.  One wave owns 16 query rows; key blocks of 32.
// ---------------------------------------------------------------------------
__global__ void __launch_bounds__(128) flash_attn_kernel(
    const _Float16* __restrict__ Qh, const _Float16* __restrict__ Kh,
    const unsigned* __restrict__ Vp, float* __restrict__ Out) {
  __shared__ _Float16 pshare[4][16][32];  // per-wave P tile (16 q x 32 keys)

  const int wave = threadIdx.x >> 5;
  const int lane = threadIdx.x & 31;
  const int tile = blockIdx.x * 4 + wave;      // over B * S/16
  const int b    = tile / (SS / 16);
  const int q0   = (tile % (SS / 16)) * 16;
  const int M    = lane & 15;
  const int half = lane >> 4;

  const _Float16* Qb = Qh + (size_t)b * SS * DK;
  const _Float16* Kb = Kh + (size_t)b * SS * DK;

  // Q A-fragments: two 16B chunks per K=32 chunk (contiguous in [S,64] layout)
  const _Float16* qrow = Qb + (size_t)(q0 + M) * DK;
  v16h qa[2];
#pragma unroll
  for (int c = 0; c < 2; ++c) {
    const uint4 lo = *(const uint4*)(const void*)(qrow + c * 32 + (half << 4));
    const uint4 hi = *(const uint4*)(const void*)(qrow + c * 32 + 8 + (half << 4));
    qa[c] = join16(lo, hi);
  }

  v8f   o[4] = {};
  float mi[8], li[8];
#pragma unroll
  for (int r = 0; r < 8; ++r) { mi[r] = -3.0e38f; li[r] = 0.0f; }

  for (int j0 = 0; j0 <= q0; j0 += 32) {
    if (j0 + 32 <= q0) {  // prefetch next K block + V fragment block to near $
      __builtin_prefetch((const void*)(Kb + (size_t)(j0 + 32 + M) * DK), 0, 3);
      __builtin_prefetch(
          (const void*)(Vp + ((size_t)((b * SS + j0 + 32) >> 5) * 4 * 32 + lane) * 8), 0, 3);
    }

    // Preload K^T fragments (4) and pre-swizzled V fragments (4); V loads
    // hide behind score WMMAs + softmax.
    v16h kf[4];  // [nt][c]
#pragma unroll
    for (int nt = 0; nt < 2; ++nt) {
      const _Float16* krow = Kb + (size_t)(j0 + (nt << 4) + M) * DK;
#pragma unroll
      for (int c = 0; c < 2; ++c)
        kf[nt * 2 + c] = *(const v16h*)(const void*)(krow + c * 32 + (half << 4));
    }
    const unsigned* vblk = Vp + (size_t)((b * SS + j0) >> 5) * 4 * 32 * 8 + lane * 8;
    v16h vf[4];
#pragma unroll
    for (int nt = 0; nt < 4; ++nt)
      vf[nt] = *(const v16h*)(const void*)(vblk + nt * 32 * 8);

    // ---- S = Q (16x64) @ K^T (64x32) ----
    v8f s[2] = {};
#pragma unroll
    for (int nt = 0; nt < 2; ++nt)
#pragma unroll
      for (int c = 0; c < 2; ++c)
        s[nt] = WMMA_F16(qa[c], kf[nt * 2 + c], s[nt]);

    // ---- causal mask (diagonal blocks only) ----
    if (j0 + 31 > q0) {
#pragma unroll
      for (int nt = 0; nt < 2; ++nt) {
        const int key = j0 + (nt << 4) + M;
#pragma unroll
        for (int r = 0; r < 8; ++r)
          if (key > q0 + r + (half << 3)) s[nt][r] = -3.0e38f;
      }
    }

    // ---- online softmax ----
    float mnew[8], psc[8];
#pragma unroll
    for (int r = 0; r < 8; ++r) {
      const float v = rowmax16(fmaxf(s[0][r], s[1][r]));
      mnew[r] = fmaxf(mi[r], v);
      psc[r]  = __builtin_amdgcn_exp2f((mi[r] - mnew[r]) * 1.44269504f);
      mi[r]   = mnew[r];
    }
#pragma unroll
    for (int nt = 0; nt < 2; ++nt)
#pragma unroll
      for (int r = 0; r < 8; ++r)
        s[nt][r] = __builtin_amdgcn_exp2f((s[nt][r] - mnew[r]) * 1.44269504f);
#pragma unroll
    for (int r = 0; r < 8; ++r)
      li[r] = li[r] * psc[r] + rowsum16(s[0][r] + s[1][r]);
#pragma unroll
    for (int nt = 0; nt < 4; ++nt)
#pragma unroll
      for (int r = 0; r < 8; ++r) o[nt][r] *= psc[r];

    // ---- P: C layout -> A fragment via per-wave LDS (ds_load_b128 reads) ---
#pragma unroll
    for (int nt = 0; nt < 2; ++nt)
#pragma unroll
      for (int r = 0; r < 8; ++r)
        pshare[wave][r + (half << 3)][(nt << 4) + M] = (_Float16)s[nt][r];
    const uint4 plo = *(const uint4*)(const void*)&pshare[wave][M][(half << 4)];
    const uint4 phi = *(const uint4*)(const void*)&pshare[wave][M][8 + (half << 4)];
    const v16h pa = join16(plo, phi);

    // ---- O += P (16x32) @ V (32x64): V fragments already in regs ----
#pragma unroll
    for (int nt = 0; nt < 4; ++nt) o[nt] = WMMA_F16(pa, vf[nt], o[nt]);
  }

  // ---- normalize + store f32 output ----
#pragma unroll
  for (int nt = 0; nt < 4; ++nt)
#pragma unroll
    for (int r = 0; r < 8; ++r) {
      const int q = q0 + r + (half << 3);
      Out[((size_t)b * SS + q) * DK + (nt << 4) + M] = o[nt][r] / li[r];
    }
}

// ---------------------------------------------------------------------------
extern "C" void kernel_launch(void* const* d_in, const int* in_sizes, int n_in,
                              void* d_out, int out_size, void* d_ws, size_t ws_size,
                              hipStream_t stream) {
  const float* X  = (const float*)d_in[0];
  const float* Wq = (const float*)d_in[1];
  const float* Wk = (const float*)d_in[2];
  const float* Wv = (const float*)d_in[3];
  float* out = (float*)d_out;

  const size_t elems = (size_t)BB * SS * DK;          // 524288
  _Float16* Qh = (_Float16*)d_ws;                     // 1 MB
  _Float16* Kh = Qh + elems;                          // 1 MB
  unsigned* Vp = (unsigned*)(Kh + elems);             // 1 MB (packed V, u32 pairs)
  unsigned* Wp = Vp + elems / 2;                      // 384 KB (packed W)

  pack_w_kernel<<<(3 * 32768) / 256, 256, 0, stream>>>(Wq, Wk, Wv, Wp);

  const int blocks = (BB * SS / 16) / 4;              // 128 blocks, 4 waves each
  qkv_proj_kernel<<<blocks, 128, 0, stream>>>(X, Wp, Qh, Kh, (_Float16*)Vp);
  flash_attn_kernel<<<blocks, 128, 0, stream>>>(Qh, Kh, Vp, out);
}
